// CrossATT_6047313952905
// MI455X (gfx1250) — matrix operand order repaired and checked by reading, not compiled, not run on hardware
//
#include <hip/hip_runtime.h>
#include <hip/hip_bf16.h>

// ---- problem constants (match reference) ----
#define B_DIM 16
#define CLEN  4096
#define QLEN  512
#define DDIM  128

// ---- LDS layout (units: _Float16 elements), padded strides to avoid bank conflicts ----
#define SK_STRIDE 136   // sK : 64 q-rows x 128 d   (row-major, B-frags for S)
#define VT_STRIDE 72    // sVt: 128 d-rows x 64 q   (transposed, B-frags for P@V)
#define W_STRIDE  136   // sW*: 128 dout-rows x 128 din (transposed, B-frags for @W)
#define P_STRIDE  136   // sP : per-wave 16 rows x 128 (A-frag relayout scratch)

#define OFF_K  0
#define OFF_VT (OFF_K  + 64  * SK_STRIDE)
#define OFF_W0 (OFF_VT + 128 * VT_STRIDE)
#define OFF_W1 (OFF_W0 + 128 * W_STRIDE)
#define OFF_P  (OFF_W1 + 128 * W_STRIDE)
#define SMEM_TOTAL (OFF_P + 8 * 16 * P_STRIDE)   // halves

typedef __attribute__((ext_vector_type(16))) _Float16 v16h;
typedef __attribute__((ext_vector_type(8)))  _Float16 v8h;
typedef __attribute__((ext_vector_type(8)))  float    v8f;
typedef __attribute__((ext_vector_type(4)))  float    v4f;

static __device__ __forceinline__ v8f zero8() {
    v8f z;
#pragma unroll
    for (int i = 0; i < 8; ++i) z[i] = 0.0f;
    return z;
}

// B-fragment (K=32 x N=16, f16): two 8-half runs at p and p+16 (K and K+16 groups)
static __device__ __forceinline__ v16h ld_b_frag(const _Float16* p) {
    v8h a = *(const v8h*)p;
    v8h b = *(const v8h*)(p + 16);
    v16h r;
#pragma unroll
    for (int i = 0; i < 8; ++i) { r[i] = a[i]; r[i + 8] = b[i]; }
    return r;
}

// A-fragment (M=16 x K=32, f16): two disjoint 8-half runs (K and K+16 groups)
static __device__ __forceinline__ v16h ld_a_frag(const _Float16* p0, const _Float16* p1) {
    v8h a = *(const v8h*)p0;
    v8h b = *(const v8h*)p1;
    v16h r;
#pragma unroll
    for (int i = 0; i < 8; ++i) { r[i] = a[i]; r[i + 8] = b[i]; }
    return r;
}

// reductions across the 16-lane half-groups (wave32: xor 1,2,4,8 stays in-half)
static __device__ __forceinline__ float halfmax16(float v) {
    v = fmaxf(v, __shfl_xor(v, 1, 32));
    v = fmaxf(v, __shfl_xor(v, 2, 32));
    v = fmaxf(v, __shfl_xor(v, 4, 32));
    v = fmaxf(v, __shfl_xor(v, 8, 32));
    return v;
}
static __device__ __forceinline__ float halfsum16(float v) {
    v += __shfl_xor(v, 1, 32);
    v += __shfl_xor(v, 2, 32);
    v += __shfl_xor(v, 4, 32);
    v += __shfl_xor(v, 8, 32);
    return v;
}

#define L2E 1.44269504088896340f

__global__ __launch_bounds__(256) void cross_att_flash_kernel(
    const float* __restrict__ xc,   // [B, CLEN, D]
    const float* __restrict__ xq,   // [B, QLEN, D]
    const float* __restrict__ W0,   // [D, D]
    const float* __restrict__ W1,   // [D, D]
    float* __restrict__ out)        // [B, CLEN, D]
{
    extern __shared__ _Float16 smem[];
    _Float16* sK  = smem + OFF_K;
    _Float16* sVt = smem + OFF_VT;
    _Float16* sW0 = smem + OFF_W0;
    _Float16* sW1 = smem + OFF_W1;
    _Float16* sP  = smem + OFF_P;

    const int tid  = threadIdx.x;
    const int lane = tid & 31;
    const int wid  = tid >> 5;
    const int lnq  = lane & 15;   // N / column-of-16 index
    const int hi   = lane >> 4;   // lane half (selects K/M sub-group)

    const int b      = blockIdx.y;
    const int c_base = blockIdx.x * 128;

    // ---- stage W0/W1 once: f32 row-major -> f16 transposed [dout][din] ----
    for (int idx = tid; idx < DDIM * DDIM; idx += 256) {
        const int i = idx >> 7, j = idx & 127;
        sW0[j * W_STRIDE + i] = (_Float16)W0[idx];
        sW1[j * W_STRIDE + i] = (_Float16)W1[idx];
    }

    // ---- x_cont rows -> A fragments in registers (K = 0..127, 4 frags) ----
    const int    myrow = c_base + wid * 16 + lnq;
    const float* xrow  = xc + ((size_t)b * CLEN + myrow) * DDIM;
    v16h a_frag[4];
#pragma unroll
    for (int k = 0; k < 4; ++k) {
        const float* p = xrow + k * 32 + hi * 8;
        v4f u0 = *(const v4f*)(p);
        v4f u1 = *(const v4f*)(p + 4);
        v4f u2 = *(const v4f*)(p + 16);
        v4f u3 = *(const v4f*)(p + 20);
        v16h r;
#pragma unroll
        for (int i = 0; i < 4; ++i) {
            r[i]      = (_Float16)u0[i];
            r[i + 4]  = (_Float16)u1[i];
            r[i + 8]  = (_Float16)u2[i];
            r[i + 12] = (_Float16)u3[i];
        }
        a_frag[k] = r;
    }

    // ---- online-softmax state ----
    v8f o_acc[8];
#pragma unroll
    for (int t = 0; t < 8; ++t) o_acc[t] = zero8();
    float m_st[8], l_st[8];
#pragma unroll
    for (int r = 0; r < 8; ++r) { m_st[r] = -3.0e38f; l_st[r] = 0.0f; }

    _Float16*    sPw  = sP + wid * 16 * P_STRIDE;
    const float* xq_b = xq + (size_t)b * QLEN * DDIM;

    // ================= main loop over 64-wide key/value chunks =================
    for (int q0 = 0; q0 < QLEN; q0 += 64) {
        __syncthreads();
        // stage chunk into both layouts (f32 -> f16)
        for (int idx = tid; idx < 64 * DDIM; idx += 256) {
            const int q = idx >> 7, d = idx & 127;
            const float    v = xq_b[(size_t)(q0 + q) * DDIM + d];
            const _Float16 h = (_Float16)v;
            sK [q * SK_STRIDE + d] = h;
            sVt[d * VT_STRIDE + q] = h;
        }
        if (q0 + 64 < QLEN)   // pull next chunk toward L2 (global_prefetch_b8)
            __builtin_prefetch(xq_b + (size_t)(q0 + 64) * DDIM + tid * 32, 0, 1);
        __syncthreads();

        // ---- S = Xc * Xq^T : 4 q-tiles of 16, K = 128 ----
        v8f s[4];
#pragma unroll
        for (int j = 0; j < 4; ++j) {
            v8f acc = zero8();
#pragma unroll
            for (int k = 0; k < 4; ++k) {
                v16h bf = ld_b_frag(sK + (j * 16 + lnq) * SK_STRIDE + k * 32 + hi * 8);
                acc = __builtin_amdgcn_wmma_f32_16x16x32_f16(
                    false, a_frag[k], false, bf, (short)0, acc, false, false);
            }
            s[j] = acc;
        }

        // ---- online softmax over this chunk ----
#pragma unroll
        for (int r = 0; r < 8; ++r) {
            float mx = fmaxf(fmaxf(s[0][r], s[1][r]), fmaxf(s[2][r], s[3][r]));
            mx = halfmax16(mx);
            const float mnew = fmaxf(m_st[r], mx);
            const float corr = exp2f((m_st[r] - mnew) * L2E);
            float rs = 0.0f;
#pragma unroll
            for (int j = 0; j < 4; ++j) {
                const float p = exp2f((s[j][r] - mnew) * L2E);
                s[j][r] = p;
                rs += p;
            }
            rs = halfsum16(rs);
            l_st[r] = l_st[r] * corr + rs;
            m_st[r] = mnew;
#pragma unroll
            for (int t = 0; t < 8; ++t) o_acc[t][r] *= corr;
        }

        // ---- relayout P (C-layout f32 -> A-layout f16) via per-wave LDS ----
#pragma unroll
        for (int j = 0; j < 4; ++j)
#pragma unroll
            for (int r = 0; r < 8; ++r)
                sPw[(r + 8 * hi) * P_STRIDE + j * 16 + lnq] = (_Float16)s[j][r];

        v16h p_frag[2];
#pragma unroll
        for (int kq = 0; kq < 2; ++kq) {
            const _Float16* p0 = sPw + lnq * P_STRIDE + kq * 32 + hi * 8;
            p_frag[kq] = ld_a_frag(p0, p0 + 16);
        }

        // ---- O += P * V (8 d-tiles, K = 64) ----
#pragma unroll
        for (int t = 0; t < 8; ++t) {
#pragma unroll
            for (int kq = 0; kq < 2; ++kq) {
                v16h vf = ld_b_frag(sVt + (t * 16 + lnq) * VT_STRIDE + kq * 32 + hi * 8);
                o_acc[t] = __builtin_amdgcn_wmma_f32_16x16x32_f16(
                    false, p_frag[kq], false, vf, (short)0, o_acc[t], false, false);
            }
        }
    }

    // ---- normalize: c2q = O / l ----
#pragma unroll
    for (int r = 0; r < 8; ++r) {
        const float inv = 1.0f / l_st[r];
#pragma unroll
        for (int t = 0; t < 8; ++t) o_acc[t][r] *= inv;
    }

    // ---- relayout c2q -> A fragments (K = 128) ----
#pragma unroll
    for (int t = 0; t < 8; ++t)
#pragma unroll
        for (int r = 0; r < 8; ++r)
            sPw[(r + 8 * hi) * P_STRIDE + t * 16 + lnq] = (_Float16)o_acc[t][r];

    v16h c_frag[4];
#pragma unroll
    for (int k = 0; k < 4; ++k) {
        const _Float16* p0 = sPw + lnq * P_STRIDE + k * 32 + hi * 8;
        c_frag[k] = ld_a_frag(p0, p0 + 16);
    }

    // ---- out = c2q @ W1 + x_cont @ W0 ----
    float* outp = out + ((size_t)b * CLEN + c_base + wid * 16) * DDIM;
#pragma unroll
    for (int t = 0; t < 8; ++t) {
        v8f f = zero8();
#pragma unroll
        for (int k = 0; k < 4; ++k) {
            v16h wf = ld_b_frag(sW1 + (t * 16 + lnq) * W_STRIDE + k * 32 + hi * 8);
            f = __builtin_amdgcn_wmma_f32_16x16x32_f16(
                false, c_frag[k], false, wf, (short)0, f, false, false);
        }
#pragma unroll
        for (int k = 0; k < 4; ++k) {
            v16h wf = ld_b_frag(sW0 + (t * 16 + lnq) * W_STRIDE + k * 32 + hi * 8);
            f = __builtin_amdgcn_wmma_f32_16x16x32_f16(
                false, a_frag[k], false, wf, (short)0, f, false, false);
        }
#pragma unroll
        for (int r = 0; r < 8; ++r)
            outp[(size_t)(r + 8 * hi) * DDIM + t * 16 + lnq] = f[r];
    }
}

extern "C" void kernel_launch(void* const* d_in, const int* in_sizes, int n_in,
                              void* d_out, int out_size, void* d_ws, size_t ws_size,
                              hipStream_t stream) {
    (void)in_sizes; (void)n_in; (void)out_size; (void)d_ws; (void)ws_size;
    const float* xc = (const float*)d_in[0];   // x_cont [B,CLEN,D] f32
    const float* xq = (const float*)d_in[1];   // x_ques [B,QLEN,D] f32
    // d_in[2] c_mask, d_in[3] q_mask: unused by the reference computation
    const float* W0 = (const float*)d_in[4];
    const float* W1 = (const float*)d_in[5];
    float* outp = (float*)d_out;

    const size_t smem_bytes = (size_t)SMEM_TOTAL * sizeof(_Float16);
    (void)hipFuncSetAttribute((const void*)cross_att_flash_kernel,
                              hipFuncAttributeMaxDynamicSharedMemorySize,
                              (int)smem_bytes);

    dim3 grid(CLEN / 128, B_DIM);
    dim3 block(256);
    hipLaunchKernelGGL(cross_att_flash_kernel, grid, block, smem_bytes, stream,
                       xc, xq, W0, W1, outp);
}
